// MQBlock_39797166964973
// MI455X (gfx1250) — matrix operand order repaired and checked by reading, not compiled, not run on hardware
//
#include <hip/hip_runtime.h>
#include <hip/hip_bf16.h>

typedef __attribute__((ext_vector_type(16))) _Float16 v16h;
typedef __attribute__((ext_vector_type(8)))  _Float16 h8;
typedef __attribute__((ext_vector_type(4)))  _Float16 h4;
typedef __attribute__((ext_vector_type(8)))  float    v8f;
typedef __attribute__((ext_vector_type(4)))  unsigned int v4u;
typedef __attribute__((ext_vector_type(8)))  int      v8i;
typedef __attribute__((ext_vector_type(4)))  int      v4i;

#define BN 16384          // rows (16*1024)
#define C  512            // channels
#define M  8192           // codes
#define NCHUNK 512        // M / 16

#if defined(__gfx1250__) && __has_builtin(__builtin_amdgcn_tensor_load_to_lds) && \
    __has_builtin(__builtin_amdgcn_s_wait_tensorcnt)
#define USE_TDM 1
#else
#define USE_TDM 0
#endif

// ---------------- counts init ----------------
__global__ void vq_zero_counts(int* __restrict__ counts) {
    int i = blockIdx.x * 256 + threadIdx.x;
    if (i < M) counts[i] = 0;
}

// ---------------- f32 -> f16 convert (x) ----------------
__global__ void vq_convert_x(const float* __restrict__ x, _Float16* __restrict__ xh) {
    int row  = blockIdx.x * 8 + (threadIdx.x >> 5);
    int lane = threadIdx.x & 31;
    const float4* src = (const float4*)(x + (size_t)row * C);
    h4* dst = (h4*)(xh + (size_t)row * C);
#pragma unroll
    for (int j = 0; j < 4; ++j) {
        float4 f = src[lane + 32 * j];
        h4 h = { (_Float16)f.x, (_Float16)f.y, (_Float16)f.z, (_Float16)f.w };
        dst[lane + 32 * j] = h;
    }
}

// ---------------- f32 -> f16 convert (embedding) + row norms ----------------
__global__ void vq_convert_e(const float* __restrict__ e, _Float16* __restrict__ eh,
                             float* __restrict__ knorm) {
    int row  = blockIdx.x * 8 + (threadIdx.x >> 5);
    int lane = threadIdx.x & 31;
    const float4* src = (const float4*)(e + (size_t)row * C);
    h4* dst = (h4*)(eh + (size_t)row * C);
    float s = 0.0f;
#pragma unroll
    for (int j = 0; j < 4; ++j) {
        float4 f = src[lane + 32 * j];
        s += f.x * f.x + f.y * f.y + f.z * f.z + f.w * f.w;
        h4 h = { (_Float16)f.x, (_Float16)f.y, (_Float16)f.z, (_Float16)f.w };
        dst[lane + 32 * j] = h;
    }
#pragma unroll
    for (int m = 16; m >= 1; m >>= 1) s += __shfl_xor(s, m, 32);
    if (lane == 0) knorm[row] = s;
}

#if USE_TDM
// Issue a 1-D TDM copy: 16KB (4096 dwords) from global srcAddr -> LDS ldsOff.
// D# layout per CDNA5 ISA ch.8: group0 {count,lds_addr,global_addr,type=2},
// group1 {data_size=4B, tensor_dim0=tile_dim0=4096, dims1=1, stride0=4096}.
// This toolchain's builtin takes 6 args (4 descriptor groups + spare group + cpol).
__device__ __forceinline__ void tdm_load_chunk(unsigned int ldsOff, unsigned long long srcAddr) {
    v4u g0;
    g0.x = 1u;                                   // count=1 (valid descriptor)
    g0.y = ldsOff;                               // lds_addr, bits 63:32
    g0.z = (unsigned int)(srcAddr & 0xffffffffu);// global_addr[31:0]
    g0.w = (unsigned int)(srcAddr >> 32) | (2u << 30); // global_addr[56:32] | type=2
    v8i g1;
    g1[0] = 0x00020000;   // workgroup_mask=0, data_size=2 (4 bytes), no flags
    g1[1] = 0x10000000;   // tensor_dim0[15:0]=4096 in bits 63:48
    g1[2] = 0x00010000;   // tensor_dim1=1 in bits 111:80
    g1[3] = 0x10000000;   // tile_dim0=4096 in bits 127:112
    g1[4] = 0x00000001;   // tile_dim1=1, tile_dim2=0
    g1[5] = 4096;         // tensor_dim0_stride[31:0]
    g1[6] = 0x10000000;   // tensor_dim1_stride[15:0]=4096 in bits 223:208
    g1[7] = 0;
    v4i gz4 = {0, 0, 0, 0};
    v8i gz8 = {0, 0, 0, 0, 0, 0, 0, 0};
    __builtin_amdgcn_tensor_load_to_lds(g0, g1, gz4, gz4, gz8, 0);
}
#endif

// ---------------- fused WMMA distance GEMM + argmin ----------------
// 128 threads = 4 waves; each wave owns a 16-row tile (64 rows / block, 256 blocks).
// A tile (16x512 f16) lives in 128 VGPRs per wave. B chunks (16 codes x 512 f16,
// contiguous 16KB) are double-buffered through LDS, fetched by the TDM engine.
__global__ void __launch_bounds__(128, 1)
vq_argmin(const _Float16* __restrict__ xh, const _Float16* __restrict__ eh,
          const float* __restrict__ knorm, int* __restrict__ idx,
          float* __restrict__ idxf, int* __restrict__ counts) {
    __shared__ alignas(64) _Float16 Bsh[2][16 * C];

    const int tid  = threadIdx.x;
    const int wave = tid >> 5;
    const int lane = tid & 31;
    const int l15  = lane & 15;
    const int hi   = lane >> 4;                 // 0 = lanes 0-15, 1 = lanes 16-31
    const int rowbase = blockIdx.x * 64 + wave * 16;

    // ---- load A tile into registers (contiguous k = 16*hi + s per 32-wide K-chunk, same order as B) ----
    v16h a[16];
    const char* xb = (const char*)xh;
    {
        const size_t rb = (size_t)(rowbase + l15) * (C * 2);
#pragma unroll
        for (int kk = 0; kk < 16; ++kk) {
            size_t base = rb + kk * 64 + hi * 32;
            h8 lo = *(const h8*)(xb + base);
            h8 hh = *(const h8*)(xb + base + 16);
            a[kk] = __builtin_shufflevector(lo, hh, 0,1,2,3,4,5,6,7,8,9,10,11,12,13,14,15);
        }
    }

    float minv[8]; int mini[8];
#pragma unroll
    for (int i = 0; i < 8; ++i) { minv[i] = 3.4e38f; mini[i] = 0; }

#if USE_TDM
    const unsigned int ldsBase = (unsigned int)(unsigned long long)(uintptr_t)&Bsh[0][0];
    const unsigned long long ebase = (unsigned long long)(uintptr_t)eh;
    if (wave == 0) {
        tdm_load_chunk(ldsBase, ebase);          // preload chunk 0 -> buf 0
        __builtin_amdgcn_s_wait_tensorcnt(0);
    }
    __syncthreads();
#else
#pragma unroll
    for (int t = 0; t < 8; ++t) {
        int u = tid + t * 128;
        float4 v = *(const float4*)((const char*)eh + (size_t)u * 16);
        *(float4*)((char*)&Bsh[0][0] + (size_t)u * 16) = v;
    }
    __syncthreads();
#endif

    for (int c = 0; c < NCHUNK; ++c) {
        const int cur = c & 1, nxt = cur ^ 1;

#if USE_TDM
        // End-of-iteration barrier of c-1 guarantees buf[nxt] is no longer read.
        if (wave == 0 && c + 1 < NCHUNK)
            tdm_load_chunk(ldsBase + (unsigned int)nxt * (16 * C * 2),
                           ebase + (unsigned long long)(c + 1) * (16 * C * 2));
#else
        float4 pf[8];
        if (c + 1 < NCHUNK) {
#pragma unroll
            for (int t = 0; t < 8; ++t) {
                int u = tid + t * 128;
                pf[t] = *(const float4*)((const char*)eh +
                         (size_t)(c + 1) * (16 * C * 2) + (size_t)u * 16);
            }
        }
#endif

        // ---- full K=512 dot for this 16x16 tile: two independent WMMA chains ----
        const float kn = knorm[c * 16 + l15];
        v8f acc0 = {}, acc1 = {};
        const char* bsh = (const char*)&Bsh[cur][0];
#pragma unroll
        for (int kk = 0; kk < 8; ++kk) {
            const v16h b0 = *(const v16h*)(bsh + l15 * (C * 2) + kk * 64 + hi * 32);
            const v16h b1 = *(const v16h*)(bsh + l15 * (C * 2) + (kk + 8) * 64 + hi * 32);
            acc0 = __builtin_amdgcn_wmma_f32_16x16x32_f16(
                false, a[kk], false, b0, (short)0, acc0, false, false);
            acc1 = __builtin_amdgcn_wmma_f32_16x16x32_f16(
                false, a[kk + 8], false, b1, (short)0, acc1, false, false);
        }

        // dist = ||k||^2 - 2 q.k   (||q||^2 constant over codes -> argmin-invariant)
#pragma unroll
        for (int i = 0; i < 8; ++i) {
            float d = kn - 2.0f * (acc0[i] + acc1[i]);
            if (d < minv[i]) { minv[i] = d; mini[i] = c * 16 + l15; }
        }

#if USE_TDM
        if (wave == 0 && c + 1 < NCHUNK)
            __builtin_amdgcn_s_wait_tensorcnt(0);
        __syncthreads();
#else
        __syncthreads();
        if (c + 1 < NCHUNK) {
#pragma unroll
            for (int t = 0; t < 8; ++t) {
                int u = tid + t * 128;
                *(float4*)((char*)&Bsh[nxt][0] + (size_t)u * 16) = pf[t];
            }
        }
        __syncthreads();
#endif
    }

    // ---- per-row min across the 16 N-lanes (keep lowest index on ties) ----
#pragma unroll
    for (int i = 0; i < 8; ++i) {
        float v = minv[i]; int ix = mini[i];
#pragma unroll
        for (int m = 8; m >= 1; m >>= 1) {
            float ov = __shfl_xor(v, m, 32);
            int   oi = __shfl_xor(ix, m, 32);
            if (ov < v || (ov == v && oi < ix)) { v = ov; ix = oi; }
        }
        if (l15 == 0) {
            int r = rowbase + i + hi * 8;
            idx[r]  = ix;
            idxf[r] = (float)ix;
            atomicAdd(&counts[ix], 1);
        }
    }
}

// ---------------- straight-through gather: z_st = q + (z - q) ----------------
__global__ void vq_gather(const float* __restrict__ x, const float* __restrict__ e,
                          const int* __restrict__ idx, float* __restrict__ zst) {
    size_t gid = (size_t)blockIdx.x * 256 + threadIdx.x;
    int n  = (int)(gid >> 7);
    int c4 = (int)(gid & 127);
    float4 q  = ((const float4*)x)[gid];
    float4 zv = ((const float4*)(e + (size_t)idx[n] * C))[c4];
    float4 r;
    r.x = q.x + (zv.x - q.x); r.y = q.y + (zv.y - q.y);
    r.z = q.z + (zv.z - q.z); r.w = q.w + (zv.w - q.w);
    ((float4*)zst)[gid] = r;
}

// ---------------- one-hot fill (512 MB, bandwidth-bound) ----------------
__global__ void vq_onehot(const int* __restrict__ idx, float* __restrict__ oh) {
    size_t gid = (size_t)blockIdx.x * 256 + threadIdx.x;
    int n  = (int)(gid >> 11);
    int m0 = (int)(gid & 2047) << 2;
    int id = idx[n];
    float4 o;
    o.x = (m0 + 0 == id) ? 1.0f : 0.0f;
    o.y = (m0 + 1 == id) ? 1.0f : 0.0f;
    o.z = (m0 + 2 == id) ? 1.0f : 0.0f;
    o.w = (m0 + 3 == id) ? 1.0f : 0.0f;
    ((float4*)oh)[gid] = o;
}

// ---------------- perplexity ----------------
__global__ void vq_perplexity(const int* __restrict__ counts, float* __restrict__ out) {
    __shared__ float red[256];
    float s = 0.0f;
    for (int j = 0; j < M / 256; ++j) {
        float p = (float)counts[threadIdx.x + 256 * j] * (1.0f / (float)BN);
        s += p * logf(p + 1e-10f);
    }
    red[threadIdx.x] = s;
    __syncthreads();
    for (int w = 128; w >= 1; w >>= 1) {
        if (threadIdx.x < w) red[threadIdx.x] += red[threadIdx.x + w];
        __syncthreads();
    }
    if (threadIdx.x == 0) out[0] = expf(-red[0]);
}

extern "C" void kernel_launch(void* const* d_in, const int* in_sizes, int n_in,
                              void* d_out, int out_size, void* d_ws, size_t ws_size,
                              hipStream_t stream) {
    (void)in_sizes; (void)n_in; (void)out_size; (void)ws_size;
    const float* x   = (const float*)d_in[0];   // [16,1024,512] f32
    const float* emb = (const float*)d_in[1];   // [8192,512]    f32
    float* out = (float*)d_out;

    // workspace layout (bytes): counts | idx | knorm | xh(f16) | eh(f16)  ~24.2 MB
    char* ws = (char*)d_ws;
    int*      counts = (int*)ws;                                  // 32 KB
    int*      idx    = (int*)(ws + 32768);                        // 64 KB
    float*    knorm  = (float*)(ws + 98304);                      // 32 KB
    _Float16* xh     = (_Float16*)(ws + 131072);                  // 16 MB
    _Float16* eh     = (_Float16*)(ws + 131072 + 16777216);       // 8 MB

    // output layout: z_st | z_indices(as f32) | z_onehot | perplexity
    float* zst  = out;
    float* idxf = out + (size_t)BN * C;
    float* oh   = out + (size_t)BN * C + BN;
    float* perp = out + (size_t)BN * C + BN + (size_t)BN * M;

    vq_zero_counts<<<(M + 255) / 256, 256, 0, stream>>>(counts);
    vq_convert_x  <<<BN / 8, 256, 0, stream>>>(x, xh);
    vq_convert_e  <<<M / 8, 256, 0, stream>>>(emb, eh, knorm);
    vq_argmin     <<<BN / 64, 128, 0, stream>>>(xh, eh, knorm, idx, idxf, counts);
    vq_gather     <<<(BN * C / 4) / 256, 256, 0, stream>>>(x, emb, idx, zst);
    vq_onehot     <<<((size_t)BN * M / 4) / 256, 256, 0, stream>>>(idx, oh);
    vq_perplexity <<<1, 256, 0, stream>>>(counts, perp);
}
